// GATBlock_4887672783237
// MI455X (gfx1250) — matrix operand order repaired and checked by reading, not compiled, not run on hardware
//
#include <hip/hip_runtime.h>
#include <hip/hip_bf16.h>
#include <stdint.h>

#define F 128
#define H 4
#define C 32
#define NEG_SLOPE 0.2f
#define BN_EPS 1e-5f

typedef __attribute__((ext_vector_type(16))) __bf16 v16bf;
typedef __attribute__((ext_vector_type(8)))  float  v8f;

// ---------- helpers ----------
__device__ __forceinline__ unsigned fmap(float f) {
    unsigned u = __float_as_uint(f);
    return (u & 0x80000000u) ? ~u : (u | 0x80000000u);   // monotonic float->uint
}
__device__ __forceinline__ float funmap(unsigned u) {
    return (u & 0x80000000u) ? __uint_as_float(u & 0x7FFFFFFFu)
                             : __uint_as_float(~u);
}

// ---------- 0: zero a float region ----------
__global__ void zero_f32_kernel(float* __restrict__ p, long long n) {
    long long i = (long long)blockIdx.x * blockDim.x + threadIdx.x;
    long long stride = (long long)gridDim.x * blockDim.x;
    for (; i < n; i += stride) p[i] = 0.0f;
}

// ---------- 1: h = x @ W^T via bf16 WMMA, f32 accumulate ----------
// Block = 256 threads = 8 waves. Block covers a 16-row tile; wave w covers cols [16w,16w+16).
__global__ void __launch_bounds__(256)
gemm_wmma_kernel(const float* __restrict__ x, const float* __restrict__ W,
                 float* __restrict__ h, int n) {
    __shared__ float xs[16][F + 4];          // padded to dodge bank conflicts
    const int tid  = threadIdx.x;
    const int row0 = blockIdx.x * 16;

    // stage 16x128 f32 tile of x into LDS (coalesced)
    for (int i = tid; i < 16 * F; i += 256) {
        int r = i >> 7, c = i & (F - 1);
        int gr = row0 + r;
        xs[r][c] = (gr < n) ? x[(long long)gr * F + c] : 0.0f;
    }
    __syncthreads();

    const int wave = tid >> 5;               // 0..7 -> column tile
    const int lane = tid & 31;
    const int col0 = wave * 16;

    // A (16x32 bf16) layout: lane m=lane&15; lanes<16 -> Kbase 0, lanes>=16 -> Kbase 16;
    // elems 0..15 K=Kbase+0..15
    const int m      = lane & 15;
    const int aKbase = (lane < 16) ? 0 : 16;
    // B (32x16 bf16) layout: col n=lane&15; lanes<16 hold K=0..15, lanes>=16 hold K=16..31
    const int bn     = col0 + (lane & 15);
    const int bKbase = (lane < 16) ? 0 : 16;

    v8f acc = {};
    #pragma unroll
    for (int ks = 0; ks < F; ks += 32) {
        v16bf a, b;
        #pragma unroll
        for (int v = 0; v < 4; ++v) {
            int k0 = ks + aKbase + 2 * v;
            a[2 * v]     = (__bf16)xs[m][k0];
            a[2 * v + 1] = (__bf16)xs[m][k0 + 1];
            int k1 = ks + aKbase + 8 + 2 * v;
            a[8 + 2 * v]     = (__bf16)xs[m][k1];
            a[8 + 2 * v + 1] = (__bf16)xs[m][k1 + 1];
        }
        const float* wrow = W + (long long)bn * F + ks + bKbase;
        #pragma unroll
        for (int v = 0; v < 8; ++v) {
            b[2 * v]     = (__bf16)wrow[2 * v];
            b[2 * v + 1] = (__bf16)wrow[2 * v + 1];
        }
        acc = __builtin_amdgcn_wmma_f32_16x16x32_bf16(
                  false, a, false, b, (short)0, acc, false, false);
    }

    // D layout: lane<16 -> rows v (0..7); lane>=16 -> rows v+8; col = col0 + (lane&15)
    const int rbase = (lane < 16) ? 0 : 8;
    const int coln  = col0 + (lane & 15);
    #pragma unroll
    for (int v = 0; v < 8; ++v) {
        int r = row0 + rbase + v;
        if (r < n) h[(long long)r * F + coln] = acc[v];
    }
}

// ---------- 2: per-(node,head) attention dots ----------
__global__ void att_kernel(const float* __restrict__ h,
                           const float* __restrict__ att_src,
                           const float* __restrict__ att_dst,
                           float* __restrict__ a_src, float* __restrict__ a_dst, int n) {
    int i = blockIdx.x * blockDim.x + threadIdx.x;
    if (i >= n * H) return;
    int node = i >> 2, hd = i & 3;
    const float* hp = h + (long long)node * F + hd * C;
    const float* as = att_src + hd * C;
    const float* ad = att_dst + hd * C;
    float s = 0.f, d = 0.f;
    #pragma unroll
    for (int c = 0; c < C; ++c) { float v = hp[c]; s += v * as[c]; d += v * ad[c]; }
    a_src[i] = s;
    a_dst[i] = d;
}

__device__ __forceinline__ void edge_sd(const long long* ei, int eid, int nE,
                                        long long& s, long long& d) {
    if (eid < nE) { s = ei[eid]; d = ei[nE + eid]; }
    else          { s = d = (long long)(eid - nE); }
}

// ---------- 3: segment max (via monotonic u32 atomicMax) ----------
__global__ void edge_max_kernel(const long long* __restrict__ ei,
                                const float* __restrict__ a_src,
                                const float* __restrict__ a_dst,
                                unsigned* __restrict__ emax, int nE, int nTot) {
    int i = blockIdx.x * blockDim.x + threadIdx.x;
    if (i >= nTot * H) return;
    int eid = i >> 2, hd = i & 3;
    long long s, d; edge_sd(ei, eid, nE, s, d);
    float e = a_src[s * H + hd] + a_dst[d * H + hd];
    e = (e > 0.f) ? e : NEG_SLOPE * e;
    atomicMax(&emax[d * H + hd], fmap(e));
}

// ---------- 4: segment exp-sum ----------
__global__ void edge_sum_kernel(const long long* __restrict__ ei,
                                const float* __restrict__ a_src,
                                const float* __restrict__ a_dst,
                                const unsigned* __restrict__ emax,
                                float* __restrict__ denom, int nE, int nTot) {
    int i = blockIdx.x * blockDim.x + threadIdx.x;
    if (i >= nTot * H) return;
    int eid = i >> 2, hd = i & 3;
    long long s, d; edge_sd(ei, eid, nE, s, d);
    float e = a_src[s * H + hd] + a_dst[d * H + hd];
    e = (e > 0.f) ? e : NEG_SLOPE * e;
    float ex = __expf(e - funmap(emax[d * H + hd]));
    atomicAdd(&denom[d * H + hd], ex);
}

// ---------- 5: aggregate messages: one wave per edge, float4 per lane ----------
__global__ void __launch_bounds__(256)
aggregate_kernel(const long long* __restrict__ ei, const float* __restrict__ h,
                 const float* __restrict__ a_src, const float* __restrict__ a_dst,
                 const unsigned* __restrict__ emax, const float* __restrict__ denom,
                 float* __restrict__ out, int nE, int nTot) {
    int eid  = (blockIdx.x * blockDim.x + threadIdx.x) >> 5;
    int lane = threadIdx.x & 31;
    if (eid >= nTot) return;
    long long s, d; edge_sd(ei, eid, nE, s, d);
    int col = lane * 4;          // cols col..col+3 all inside head col>>5
    int hd  = col >> 5;
    float e = a_src[s * H + hd] + a_dst[d * H + hd];
    e = (e > 0.f) ? e : NEG_SLOPE * e;
    float ex    = __expf(e - funmap(emax[d * H + hd]));
    float alpha = ex / (denom[d * H + hd] + 1e-16f);
    const float4 hv = *(const float4*)(h + (long long)s * F + col);
    float* o = out + (long long)d * F + col;
    atomicAdd(o + 0, hv.x * alpha);
    atomicAdd(o + 1, hv.y * alpha);
    atomicAdd(o + 2, hv.z * alpha);
    atomicAdd(o + 3, hv.w * alpha);
}

// ---------- 6: bias + ReLU + per-column sum/sumsq ----------
__global__ void __launch_bounds__(256)
bias_relu_stats_kernel(float* __restrict__ out, const float* __restrict__ bias,
                       float* __restrict__ stats, int n) {
    int tid = threadIdx.x;
    int col = tid & (F - 1);
    int gid = blockIdx.x * blockDim.x + tid;
    int row = gid >> 7;
    int rstride = (gridDim.x * blockDim.x) >> 7;
    float b = bias[col];
    float s = 0.f, sq = 0.f;
    for (int r = row; r < n; r += rstride) {
        long long idx = (long long)r * F + col;
        float v = out[idx] + b;
        v = (v > 0.f) ? v : 0.f;
        out[idx] = v;
        s += v; sq += v * v;
    }
    __shared__ float sh[2][256];
    sh[0][tid] = s; sh[1][tid] = sq;
    __syncthreads();
    if (tid < 128) {
        atomicAdd(&stats[col],     sh[0][tid] + sh[0][tid + 128]);
        atomicAdd(&stats[F + col], sh[1][tid] + sh[1][tid + 128]);
    }
}

// ---------- 7: apply BatchNorm ----------
__global__ void bn_kernel(float* __restrict__ out, const float* __restrict__ stats,
                          const float* __restrict__ gamma, const float* __restrict__ beta,
                          int n) {
    long long i = (long long)blockIdx.x * blockDim.x + threadIdx.x;
    long long total = (long long)n * F;
    if (i >= total) return;
    int col = (int)(i & (F - 1));
    float inv_n = 1.0f / (float)n;
    float mean = stats[col] * inv_n;
    float var  = stats[F + col] * inv_n - mean * mean;
    out[i] = (out[i] - mean) * rsqrtf(var + BN_EPS) * gamma[col] + beta[col];
}

extern "C" void kernel_launch(void* const* d_in, const int* in_sizes, int n_in,
                              void* d_out, int out_size, void* d_ws, size_t ws_size,
                              hipStream_t stream) {
    const float*     x       = (const float*)d_in[0];
    const long long* ei      = (const long long*)d_in[1];   // int64 [2,E]
    const float*     W       = (const float*)d_in[2];
    const float*     att_src = (const float*)d_in[3];
    const float*     att_dst = (const float*)d_in[4];
    const float*     bias    = (const float*)d_in[5];
    const float*     gamma   = (const float*)d_in[6];
    const float*     beta    = (const float*)d_in[7];
    float*           out     = (float*)d_out;

    const int n    = in_sizes[0] / F;       // 50000
    const int nE   = in_sizes[1] / 2;       // 800000
    const int nTot = nE + n;                // edges + self loops

    // workspace layout
    char* ws = (char*)d_ws;
    float*    h      = (float*)ws;     ws += (size_t)n * F * sizeof(float);
    float*    a_src  = (float*)ws;     ws += (size_t)n * H * sizeof(float);
    float*    a_dst  = (float*)ws;     ws += (size_t)n * H * sizeof(float);
    unsigned* emax   = (unsigned*)ws;  ws += (size_t)n * H * sizeof(unsigned);
    float*    denom  = (float*)ws;     ws += (size_t)n * H * sizeof(float);
    float*    stats  = (float*)ws;     // 2*F floats

    // 0) zero accumulators: emax/denom/stats are contiguous; out separately
    long long zlen = (long long)n * H * 2 + 2 * F;   // emax + denom + stats
    zero_f32_kernel<<<512, 256, 0, stream>>>((float*)emax, zlen);
    zero_f32_kernel<<<2048, 256, 0, stream>>>(out, (long long)n * F);

    // 1) h = x @ W^T (bf16 WMMA, f32 accum)
    gemm_wmma_kernel<<<(n + 15) / 16, 256, 0, stream>>>(x, W, h, n);

    // 2) attention dots
    att_kernel<<<(n * H + 255) / 256, 256, 0, stream>>>(h, att_src, att_dst, a_src, a_dst, n);

    // 3) segment max
    edge_max_kernel<<<((long long)nTot * H + 255) / 256, 256, 0, stream>>>(
        ei, a_src, a_dst, emax, nE, nTot);

    // 4) segment exp-sum
    edge_sum_kernel<<<((long long)nTot * H + 255) / 256, 256, 0, stream>>>(
        ei, a_src, a_dst, emax, denom, nE, nTot);

    // 5) weighted aggregation (wave per edge)
    aggregate_kernel<<<(nTot + 7) / 8, 256, 0, stream>>>(
        ei, h, a_src, a_dst, emax, denom, out, nE, nTot);

    // 6) bias + ReLU + column stats
    bias_relu_stats_kernel<<<1024, 256, 0, stream>>>(out, bias, stats, n);

    // 7) BatchNorm apply
    bn_kernel<<<((long long)n * F + 255) / 256, 256, 0, stream>>>(out, stats, gamma, beta, n);
}
